// GraphSAGELayer_78451872628893
// MI455X (gfx1250) — matrix operand order repaired and checked by reading, not compiled, not run on hardware
//
#include <hip/hip_runtime.h>

// GraphSAGE fused layer for gfx1250 (MI455X), wave32, WMMA bf16 path.
// adj is a 0/1 mask -> exact in bf16. X/W use 2-term bf16 splits for ~f32 accuracy.
// B-operand staging: CDNA5 async global->LDS loads (ASYNCcnt), double-buffered
// with one barrier per stage; prefetch overlaps WMMA compute.

typedef __attribute__((ext_vector_type(16))) __bf16 v16bf;
typedef __attribute__((ext_vector_type(8)))  float  v8f;

#define NN   16384
#define DIN  64
#define DOUT 64

__device__ __forceinline__ v8f wmma_bf16f32(v16bf a, v16bf b, v8f c) {
  // D = A(16x32 bf16) * B(32x16 bf16) + C(16x16 f32)
  return __builtin_amdgcn_wmma_f32_16x16x32_bf16(false, a, false, b, (short)0, c,
                                                 false, false);
}

// ---------------------------------------------------------------------------
// prep_x: X (f32, row-major NxDIN) -> bf16 hi/lo blobs, pre-swizzled into the
// WMMA B-operand VGPR layout so the GEMM reads them with plain b128 LDS loads.
// Blob addressing: idx = ((t*4 + c)*32 + L)*16 + e
//   K = 32*t + (L>=16 ? 16 : 0) + e   (k order used here: lanes 0-15 K=0..15, 16-31 K=16..31)
//   n = 16*c + (L & 15)
// ---------------------------------------------------------------------------
__global__ __launch_bounds__(256) void prep_x_kernel(const float* __restrict__ X,
                                                     __bf16* __restrict__ hi,
                                                     __bf16* __restrict__ lo) {
  int idx = blockIdx.x * 256 + threadIdx.x;         // 0 .. N*DIN-1
  int e = idx & 15;
  int L = (idx >> 4) & 31;
  int c = (idx >> 9) & 3;
  int t = idx >> 11;
  int k = t * 32 + ((L & 16) ? 16 : 0) + e;
  int n = c * 16 + (L & 15);
  float x = X[(size_t)k * DIN + n];
  __bf16 h = (__bf16)x;
  hi[idx] = h;
  lo[idx] = (__bf16)(x - (float)h);
}

// Same swizzle for W (2*DIN x DOUT), K range 0..127 (s2 = K-tile of 32).
__global__ __launch_bounds__(256) void prep_w_kernel(const float* __restrict__ W,
                                                     __bf16* __restrict__ hi,
                                                     __bf16* __restrict__ lo) {
  int idx = blockIdx.x * 256 + threadIdx.x;         // 0 .. 8191
  int e  = idx & 15;
  int L  = (idx >> 4) & 31;
  int c  = (idx >> 9) & 3;
  int s2 = idx >> 11;
  int row = s2 * 32 + ((L & 16) ? 16 : 0) + e;      // 0..127
  int col = c * 16 + (L & 15);                      // 0..63
  float x = W[(size_t)row * DOUT + col];
  __bf16 h = (__bf16)x;
  hi[idx] = h;
  lo[idx] = (__bf16)(x - (float)h);
}

// ---------------------------------------------------------------------------
// Fused main kernel. 4 waves/block; each wave owns a 16-row tile.
// Phase 1: acc = adj @ (Xhi + Xlo)  via bf16 WMMA, adj converted in-register
//          (exact), row-sums accumulated in VALU for deg.
// Epilogue: y = (acc + X)/deg ; concat [X|y] staged to LDS (row-major, padded)
// Phase 2: z = relu(concat @ W + b) via bf16-split WMMA; L2-normalize rows
//          with shfl_xor reductions; store final output.
// LDS: two 16KB B staging buffers; concat tiles overlap them post-loop.
// ---------------------------------------------------------------------------
#define CSTRIDE 132   // padded concat row stride (floats) to avoid bank conflicts
#define NSTAGE  (NN / 64)   // 256 stages, 2 K-tiles each

__global__ __launch_bounds__(128) void sage_fused_kernel(
    const float*  __restrict__ adj, const float* __restrict__ X,
    const float*  __restrict__ bias,
    const __bf16* __restrict__ XbHi, const __bf16* __restrict__ XbLo,
    const __bf16* __restrict__ Wbh,  const __bf16* __restrict__ Wbl,
    float* __restrict__ out)
{
  // During K loop: buf0 [0,16384) (hi 8K | lo 8K), buf1 [16384,32768).
  // After final barrier: per-wave concat tiles [0, 33792).
  __shared__ __align__(16) unsigned char sm[4 * 16 * CSTRIDE * 4];  // 33792 B

  const int tid  = threadIdx.x;
  const int lane = tid & 31;
  const int wv   = tid >> 5;        // 0..3
  const int half = lane >> 4;       // 0/1
  const int l16  = lane & 15;
  const int rowTile = blockIdx.x * 4 + wv;   // 0..1023
  const int row0 = rowTile * 16;

  v8f acc[4] = {};
  float rsp0 = 0.0f, rsp1 = 0.0f, rsp2 = 0.0f, rsp3 = 0.0f;

  // A-operand base: row = row0 + l16; k order used here: lanes0-15 K 0-15,
  // lanes16-31 K 16-31 -> one per-lane contiguous chunk at +half*16.
  const float* arow = adj + (size_t)(row0 + l16) * NN + half * 16;

  // Async staging: each thread moves 4x b128 per split per stage.
  const unsigned smBase = (unsigned)(uintptr_t)(void*)(sm);
  const unsigned gOff   = (unsigned)tid * 16u;

  // Issue one stage's async batch (8 ops) into buffer at bufOff.
  auto issue_stage = [&](int t, unsigned bufOff) {
    const __bf16* gh = XbHi + (size_t)t * 4096;   // 8192 bytes per split
    const __bf16* gl = XbLo + (size_t)t * 4096;
    unsigned ldsHi = smBase + bufOff + gOff;
    unsigned ldsLo = ldsHi + 8192u;
    // CDNA5 async global->LDS: INST_OFFSET applies to both LDS and global
    // sides, so chunk strides fold into the immediate.
    asm volatile(
      "global_load_async_to_lds_b128 %0, %2, %3 offset:0\n\t"
      "global_load_async_to_lds_b128 %0, %2, %3 offset:2048\n\t"
      "global_load_async_to_lds_b128 %0, %2, %3 offset:4096\n\t"
      "global_load_async_to_lds_b128 %0, %2, %3 offset:6144\n\t"
      "global_load_async_to_lds_b128 %1, %2, %4 offset:0\n\t"
      "global_load_async_to_lds_b128 %1, %2, %4 offset:2048\n\t"
      "global_load_async_to_lds_b128 %1, %2, %4 offset:4096\n\t"
      "global_load_async_to_lds_b128 %1, %2, %4 offset:6144"
      :: "v"(ldsHi), "v"(ldsLo), "v"(gOff), "s"(gh), "s"(gl)
      : "memory");
  };

  issue_stage(0, 0u);   // prefetch stage 0 into buf0

  #pragma unroll 2
  for (int tt = 0; tt < NSTAGE; ++tt) {
    // Batch tt was issued one iteration ago -> latency already hidden.
    asm volatile("s_wait_asynccnt 0" ::: "memory");
    __syncthreads();   // all portions of buf[tt&1] visible; buf[~tt&1] free
    if (tt + 1 < NSTAGE) issue_stage(tt + 1, (unsigned)((tt + 1) & 1) * 16384u);

    const unsigned bufOff = (unsigned)(tt & 1) * 16384u;

    #pragma unroll
    for (int u = 0; u < 2; ++u) {
      const float* ap = arow + tt * 64 + u * 32;
      float4 q0 = ((const float4*)ap)[0];
      float4 q1 = ((const float4*)ap)[1];
      float4 q2 = ((const float4*)ap)[2];
      float4 q3 = ((const float4*)ap)[3];
      float vals[16] = {q0.x,q0.y,q0.z,q0.w, q1.x,q1.y,q1.z,q1.w,
                        q2.x,q2.y,q2.z,q2.w, q3.x,q3.y,q3.z,q3.w};
      v16bf a;
      #pragma unroll
      for (int e = 0; e < 16; e += 4) {      // 4 partial sums: short dep chains
        rsp0 += vals[e + 0]; rsp1 += vals[e + 1];
        rsp2 += vals[e + 2]; rsp3 += vals[e + 3];
        a[e + 0] = (__bf16)vals[e + 0];
        a[e + 1] = (__bf16)vals[e + 1];
        a[e + 2] = (__bf16)vals[e + 2];
        a[e + 3] = (__bf16)vals[e + 3];
      }

      #pragma unroll
      for (int c = 0; c < 4; ++c) {
        union { uint4 q[2]; v16bf v; } bh, bl;
        const uint4* ph =
            (const uint4*)(sm + bufOff + u * 4096 + (c * 32 + lane) * 32);
        const uint4* pl =
            (const uint4*)(sm + bufOff + 8192 + u * 4096 + (c * 32 + lane) * 32);
        bh.q[0] = ph[0]; bh.q[1] = ph[1];
        bl.q[0] = pl[0]; bl.q[1] = pl[1];
        acc[c] = wmma_bf16f32(a, bh.v, acc[c]);
        acc[c] = wmma_bf16f32(a, bl.v, acc[c]);
      }
    }
  }
  __syncthreads();   // B buffers dead; LDS reused for concat tiles below

  // degree: rowsum(adj)+1 clamped. Lane L holds partial for row (L&15).
  float rs = (rsp0 + rsp1) + (rsp2 + rsp3);
  float rsAll = rs + __shfl_xor(rs, 16);
  float deg = fmaxf(rsAll + 1.0f, 1.0f);

  // Epilogue 1: build concat = [X | (acc + X)/deg] in wave-private LDS tile.
  // C/D layout: VGPR j -> M = j + half*8, N = l16.
  float* cs = (float*)(sm + wv * (16 * CSTRIDE * 4));
  #pragma unroll
  for (int c = 0; c < 4; ++c) {
    #pragma unroll
    for (int j = 0; j < 8; ++j) {
      int m = j + half * 8;
      float dm = __shfl(deg, m);                 // deg of row m lives in lane m
      float xv = X[(size_t)(row0 + m) * DIN + c * 16 + l16];
      float y  = (acc[c][j] + xv) / dm;
      cs[m * CSTRIDE + c * 16 + l16]      = xv;  // cols 0..63  = X
      cs[m * CSTRIDE + 64 + c * 16 + l16] = y;   // cols 64..127 = h_neigh
    }
  }
  // cs is wave-private; LDS ops are in-order within a wave -> no block barrier.

  // Phase 2: z = concat @ W (bf16 split: AhWh + AhWl + AlWh), K = 128.
  v8f z[4] = {};
  #pragma unroll
  for (int s2 = 0; s2 < 4; ++s2) {
    const float* crow = cs + l16 * CSTRIDE + s2 * 32 + half * 16;  // A layout
    float4 q0 = ((const float4*)crow)[0];
    float4 q1 = ((const float4*)crow)[1];
    float4 q2 = ((const float4*)crow)[2];
    float4 q3 = ((const float4*)crow)[3];
    float vals[16] = {q0.x,q0.y,q0.z,q0.w, q1.x,q1.y,q1.z,q1.w,
                      q2.x,q2.y,q2.z,q2.w, q3.x,q3.y,q3.z,q3.w};
    v16bf ah, al;
    #pragma unroll
    for (int e = 0; e < 16; ++e) {
      __bf16 h = (__bf16)vals[e];
      ah[e] = h;
      al[e] = (__bf16)(vals[e] - (float)h);
    }
    #pragma unroll
    for (int c = 0; c < 4; ++c) {
      union { uint4 q[2]; v16bf v; } wh, wl;
      const uint4* pw = (const uint4*)(Wbh + (size_t)((s2 * 4 + c) * 32 + lane) * 16);
      const uint4* qw = (const uint4*)(Wbl + (size_t)((s2 * 4 + c) * 32 + lane) * 16);
      wh.q[0] = pw[0]; wh.q[1] = pw[1];
      wl.q[0] = qw[0]; wl.q[1] = qw[1];
      z[c] = wmma_bf16f32(ah, wh.v, z[c]);
      z[c] = wmma_bf16f32(ah, wl.v, z[c]);
      z[c] = wmma_bf16f32(al, wh.v, z[c]);
    }
  }

  // bias + relu + row L2 normalize + store
  float bc[4];
  #pragma unroll
  for (int c = 0; c < 4; ++c) bc[c] = bias[c * 16 + l16];

  #pragma unroll
  for (int j = 0; j < 8; ++j) {
    float v0 = fmaxf(z[0][j] + bc[0], 0.0f);
    float v1 = fmaxf(z[1][j] + bc[1], 0.0f);
    float v2 = fmaxf(z[2][j] + bc[2], 0.0f);
    float v3 = fmaxf(z[3][j] + bc[3], 0.0f);
    float s = v0 * v0 + v1 * v1 + v2 * v2 + v3 * v3;
    s += __shfl_xor(s, 1);
    s += __shfl_xor(s, 2);
    s += __shfl_xor(s, 4);
    s += __shfl_xor(s, 8);           // full row (64 cols) within each half
    float scale = 1.0f / fmaxf(sqrtf(s), 1e-12f);
    size_t ro = (size_t)(row0 + j + half * 8) * DOUT + l16;
    out[ro +  0] = v0 * scale;
    out[ro + 16] = v1 * scale;
    out[ro + 32] = v2 * scale;
    out[ro + 48] = v3 * scale;
  }
}

// ---------------------------------------------------------------------------
extern "C" void kernel_launch(void* const* d_in, const int* in_sizes, int n_in,
                              void* d_out, int out_size, void* d_ws, size_t ws_size,
                              hipStream_t stream) {
  const float* X   = (const float*)d_in[0];   // (N, 64)
  const float* adj = (const float*)d_in[1];   // (N, N)
  const float* W   = (const float*)d_in[2];   // (128, 64)
  const float* b   = (const float*)d_in[3];   // (64,)
  float* out = (float*)d_out;                 // (N, 64)

  // workspace layout (bf16): XbHi[N*64] | XbLo[N*64] | Wbh[128*64] | Wbl[128*64]
  __bf16* XbHi = (__bf16*)d_ws;
  __bf16* XbLo = XbHi + (size_t)NN * DIN;
  __bf16* Wbh  = XbLo + (size_t)NN * DIN;
  __bf16* Wbl  = Wbh + (size_t)2 * DIN * DOUT;

  prep_x_kernel<<<(NN * DIN) / 256, 256, 0, stream>>>(X, XbHi, XbLo);
  prep_w_kernel<<<(2 * DIN * DOUT) / 256, 256, 0, stream>>>(W, Wbh, Wbl);
  sage_fused_kernel<<<256, 128, 0, stream>>>(adj, X, b, XbHi, XbLo, Wbh, Wbl, out);
}
